// SelfAttention_70617852281426
// MI455X (gfx1250) — matrix operand order repaired and checked by reading, not compiled, not run on hardware
//
#include <hip/hip_runtime.h>

// ---------------------------------------------------------------------------
// SAGAN self-attention for MI455X (gfx1250), all matmuls via v_wmma bf16.
// ---------------------------------------------------------------------------

typedef __attribute__((ext_vector_type(16))) __bf16 v16bf;
typedef __attribute__((ext_vector_type(8)))  __bf16 bf16x8;
typedef __attribute__((ext_vector_type(4)))  __bf16 bf16x4;
typedef __attribute__((ext_vector_type(8)))  float  v8f;
typedef __attribute__((ext_vector_type(8)))  float  f32x8;
typedef __attribute__((ext_vector_type(4)))  float  f32x4;

#define DEVINL static __device__ __forceinline__

#define BB   4
#define CC   512
#define CD   64          // C/8
#define NN   4096        // H*W
#define MTOT 16384       // B*N

DEVINL v8f vzero8() {
  v8f z = {0.f, 0.f, 0.f, 0.f, 0.f, 0.f, 0.f, 0.f};
  return z;
}

DEVINL v8f wmma_bf16(v16bf a, v16bf b, v8f c) {
  // (neg_a, A, neg_b, B, c_mod, C, reuse_a, reuse_b)
  return __builtin_amdgcn_wmma_f32_16x16x32_bf16(false, a, false, b,
                                                 (short)0, c, false, false);
}

// A operand tile [16 x 32] bf16, row-major source, ld elements per row.
// Lane L: row = L&15, K chunks [kh, kh+8) and [16+kh, 16+kh+8), kh = 8*(L>=16).
DEVINL v16bf loadA(const __bf16* src, int ld, int lane) {
  int r  = lane & 15;
  int kh = (lane >> 4) << 3;
  const __bf16* p = src + (size_t)r * ld + kh;
  bf16x8 lo = *(const bf16x8*)(p);
  bf16x8 hi = *(const bf16x8*)(p + 16);
  return __builtin_shufflevector(lo, hi, 0,1,2,3,4,5,6,7,8,9,10,11,12,13,14,15);
}

// Same A layout but source is f32 (converted to bf16 on the fly).
DEVINL v16bf loadA_f32(const float* src, int ld, int lane) {
  int r  = lane & 15;
  int kh = (lane >> 4) << 3;
  const float* p = src + (size_t)r * ld + kh;
  f32x8 lo = *(const f32x8*)(p);
  f32x8 hi = *(const f32x8*)(p + 16);
  bf16x8 blo = __builtin_convertvector(lo, bf16x8);
  bf16x8 bhi = __builtin_convertvector(hi, bf16x8);
  return __builtin_shufflevector(blo, bhi, 0,1,2,3,4,5,6,7,8,9,10,11,12,13,14,15);
}

// B operand tile [32 x 16] bf16. srcT holds B transposed: srcT[c][k], ld per row.
// Lane L: column = L&15, K chunks [kh, kh+8) and [16+kh, 16+kh+8), kh = 8*(L>=16).
DEVINL v16bf loadB(const __bf16* srcT, int ld, int lane) {
  int c  = lane & 15;
  int kb = (lane >> 4) << 3;
  const __bf16* p = srcT + (size_t)c * ld + kb;
  bf16x8 lo = *(const bf16x8*)(p);
  bf16x8 hi = *(const bf16x8*)(p + 16);
  return __builtin_shufflevector(lo, hi, 0,1,2,3,4,5,6,7,8,9,10,11,12,13,14,15);
}

// ---------------------------------------------------------------------------
// K0a: x (f32) -> xb (bf16), vectorized 4-wide.
__global__ void sa_cvt_x(const float* __restrict__ x, __bf16* __restrict__ xb, int n4) {
  int i = blockIdx.x * blockDim.x + threadIdx.x;
  if (i >= n4) return;
  f32x4 v = ((const f32x4*)x)[i];
  ((bf16x4*)xb)[i] = __builtin_convertvector(v, bf16x4);
}

// K0b: W [K x N] f32 -> WT [N x K] bf16.
__global__ void sa_cvt_wT(const float* __restrict__ w, __bf16* __restrict__ wT,
                          int K, int N) {
  int i = blockIdx.x * blockDim.x + threadIdx.x;
  if (i >= K * N) return;
  int k = i / N, n = i - k * N;
  wT[(size_t)n * K + k] = (__bf16)w[i];
}

// ---------------------------------------------------------------------------
// Projection GEMM: C = A[M x K] * B[K x Ncols], B given transposed (BT[N x K]).
// Each wave computes a 16x64 tile. transBatchN != 0 stores C transposed
// per batch: C[(b*Ncols + n)*transBatchN + m%transBatchN]  (for hvT).
__global__ void sa_gemm_proj(const __bf16* __restrict__ A,
                             const __bf16* __restrict__ BT,
                             __bf16* __restrict__ Cc,
                             int M, int Ncols, int K, int transBatchN) {
  int lane = threadIdx.x & 31;
  int wid  = blockIdx.x * (blockDim.x >> 5) + (threadIdx.x >> 5);
  int tilesN = Ncols >> 6;
  int nTiles = (M >> 4) * tilesN;
  if (wid >= nTiles) return;
  int rowTile = (wid / tilesN) << 4;
  int colTile = (wid % tilesN) << 6;

  v8f acc[4] = {vzero8(), vzero8(), vzero8(), vzero8()};
  const __bf16* ap = A + (size_t)rowTile * K;
  for (int k = 0; k < K; k += 32) {
    v16bf a = loadA(ap + k, K, lane);
#pragma unroll
    for (int j = 0; j < 4; ++j) {
      v16bf b = loadB(BT + (size_t)(colTile + j * 16) * K + k, K, lane);
      acc[j] = wmma_bf16(a, b, acc[j]);
    }
  }

  int hi8 = (lane >> 4) << 3;
  int cn  = lane & 15;
#pragma unroll
  for (int j = 0; j < 4; ++j) {
#pragma unroll
    for (int v = 0; v < 8; ++v) {
      int m = rowTile + hi8 + v;
      int n = colTile + j * 16 + cn;
      __bf16 val = (__bf16)acc[j][v];
      if (transBatchN) {
        int b  = m / transBatchN;
        int ml = m - b * transBatchN;
        Cc[((size_t)b * Ncols + n) * transBatchN + ml] = val;
      } else {
        Cc[(size_t)m * Ncols + n] = val;
      }
    }
  }
}

// ---------------------------------------------------------------------------
// Attention scores + softmax -> beta (f32, written to d_out's beta region).
// One wave owns 16 query rows. Pass 1: online rowmax/rowsum over all 4096
// keys (flash-style). Pass 2: recompute s tiles, write beta = exp(s-m)/l.
__global__ void sa_attn_beta(const __bf16* __restrict__ g,
                             const __bf16* __restrict__ f,
                             float* __restrict__ beta) {
  int lane = threadIdx.x & 31;
  int wid  = blockIdx.x * (blockDim.x >> 5) + (threadIdx.x >> 5);
  if (wid >= BB * (NN >> 4)) return;
  int b    = wid >> 8;          // 256 row-tiles per batch
  int row0 = (wid & 255) << 4;

  const __bf16* gp = g + ((size_t)b * NN + row0) * CD;
  const __bf16* fp = f + (size_t)b * NN * CD;
  v16bf a0 = loadA(gp, CD, lane);        // K = 0..31
  v16bf a1 = loadA(gp + 32, CD, lane);   // K = 32..63

  float mrow[8], lrow[8];
#pragma unroll
  for (int v = 0; v < 8; ++v) { mrow[v] = -3.0e38f; lrow[v] = 0.f; }

  // pass 1: online max / sum
  for (int j = 0; j < (NN >> 4); ++j) {
    const __bf16* fr = fp + (size_t)(j << 4) * CD;
    v16bf b0 = loadB(fr, CD, lane);
    v16bf b1 = loadB(fr + 32, CD, lane);
    v8f s = wmma_bf16(a0, b0, vzero8());
    s = wmma_bf16(a1, b1, s);
#pragma unroll
    for (int v = 0; v < 8; ++v) {
      float sv = s[v];
      float mx = sv;
      mx = fmaxf(mx, __shfl_xor(mx, 1));
      mx = fmaxf(mx, __shfl_xor(mx, 2));
      mx = fmaxf(mx, __shfl_xor(mx, 4));
      mx = fmaxf(mx, __shfl_xor(mx, 8));
      float mnew = fmaxf(mrow[v], mx);
      float e = __expf(sv - mnew);
      e += __shfl_xor(e, 1);
      e += __shfl_xor(e, 2);
      e += __shfl_xor(e, 4);
      e += __shfl_xor(e, 8);
      lrow[v] = lrow[v] * __expf(mrow[v] - mnew) + e;
      mrow[v] = mnew;
    }
  }
  float rinv[8];
#pragma unroll
  for (int v = 0; v < 8; ++v) rinv[v] = 1.0f / lrow[v];

  // pass 2: recompute, normalize, write beta
  int hi8 = (lane >> 4) << 3;
  int cn  = lane & 15;
  float* bout = beta + ((size_t)b * NN + row0) * NN;
  for (int j = 0; j < (NN >> 4); ++j) {
    const __bf16* fr = fp + (size_t)(j << 4) * CD;
    v16bf b0 = loadB(fr, CD, lane);
    v16bf b1 = loadB(fr + 32, CD, lane);
    v8f s = wmma_bf16(a0, b0, vzero8());
    s = wmma_bf16(a1, b1, s);
#pragma unroll
    for (int v = 0; v < 8; ++v) {
      float p = __expf(s[v] - mrow[v]) * rinv[v];
      bout[(size_t)(hi8 + v) * NN + (j << 4) + cn] = p;
    }
  }
}

// ---------------------------------------------------------------------------
// o = beta @ hv : per batch [4096x4096]f32 @ hvT[512x4096]bf16 -> o bf16.
__global__ void sa_attn_o(const float* __restrict__ beta,
                          const __bf16* __restrict__ hvT,
                          __bf16* __restrict__ o) {
  int lane = threadIdx.x & 31;
  int wid  = blockIdx.x * (blockDim.x >> 5) + (threadIdx.x >> 5);
  if (wid >= BB * (NN >> 4) * (CC >> 6)) return;
  int b    = wid >> 11;               // 2048 tiles per batch
  int rem  = wid & 2047;
  int row0 = (rem >> 3) << 4;
  int col0 = (rem & 7) << 6;

  const float*  ap = beta + ((size_t)b * NN + row0) * NN;
  const __bf16* bp = hvT + ((size_t)b * CC + col0) * NN;

  v8f acc[4] = {vzero8(), vzero8(), vzero8(), vzero8()};
  for (int k = 0; k < NN; k += 32) {
    __builtin_prefetch(ap + k + 128, 0, 1);   // global_prefetch next A tiles
    v16bf a = loadA_f32(ap + k, NN, lane);
#pragma unroll
    for (int j = 0; j < 4; ++j) {
      v16bf bm = loadB(bp + (size_t)(j * 16) * NN + k, NN, lane);
      acc[j] = wmma_bf16(a, bm, acc[j]);
    }
  }

  int hi8 = (lane >> 4) << 3;
  int cn  = lane & 15;
#pragma unroll
  for (int j = 0; j < 4; ++j) {
#pragma unroll
    for (int v = 0; v < 8; ++v) {
      o[((size_t)(b * NN + row0 + hi8 + v)) * CC + col0 + j * 16 + cn] =
          (__bf16)acc[j][v];
    }
  }
}

// ---------------------------------------------------------------------------
// y = gamma * (o @ Wv) + x   (f32 output)
__global__ void sa_gemm_final(const __bf16* __restrict__ A,   // o bf16 [M x 512]
                              const __bf16* __restrict__ BT,  // WvT bf16 [512 x 512]
                              const float* __restrict__ x,
                              const float* __restrict__ gamma,
                              float* __restrict__ y) {
  int lane = threadIdx.x & 31;
  int wid  = blockIdx.x * (blockDim.x >> 5) + (threadIdx.x >> 5);
  if (wid >= (MTOT >> 4) * (CC >> 6)) return;
  int rowTile = (wid >> 3) << 4;
  int colTile = (wid & 7) << 6;

  v8f acc[4] = {vzero8(), vzero8(), vzero8(), vzero8()};
  const __bf16* ap = A + (size_t)rowTile * CC;
  for (int k = 0; k < CC; k += 32) {
    v16bf a = loadA(ap + k, CC, lane);
#pragma unroll
    for (int j = 0; j < 4; ++j) {
      v16bf b = loadB(BT + (size_t)(colTile + j * 16) * CC + k, CC, lane);
      acc[j] = wmma_bf16(a, b, acc[j]);
    }
  }

  float gm = gamma[0];
  int hi8 = (lane >> 4) << 3;
  int cn  = lane & 15;
#pragma unroll
  for (int j = 0; j < 4; ++j) {
#pragma unroll
    for (int v = 0; v < 8; ++v) {
      size_t idx = (size_t)(rowTile + hi8 + v) * CC + colTile + j * 16 + cn;
      y[idx] = gm * acc[j][v] + x[idx];
    }
  }
}

// ---------------------------------------------------------------------------
extern "C" void kernel_launch(void* const* d_in, const int* in_sizes, int n_in,
                              void* d_out, int out_size, void* d_ws, size_t ws_size,
                              hipStream_t stream) {
  const float* x     = (const float*)d_in[0];
  const float* Wf    = (const float*)d_in[1];
  const float* Wg    = (const float*)d_in[2];
  const float* Wh    = (const float*)d_in[3];
  const float* Wv    = (const float*)d_in[4];
  const float* gamma = (const float*)d_in[5];

  float* y    = (float*)d_out;
  float* beta = y + (size_t)MTOT * CC;   // beta region of d_out

  char* ws = (char*)d_ws;
  __bf16* xb  = (__bf16*)(ws + 0);          // 16 MB: x as bf16 [M x 512]
  __bf16* fb  = (__bf16*)(ws + 16777216);   //  2 MB: keys   [M x 64]
  __bf16* gb  = (__bf16*)(ws + 18874368);   //  2 MB: query  [M x 64]
  __bf16* hvT = (__bf16*)(ws + 20971520);   // 16 MB: value transposed [B][512 x 4096]
  __bf16* ob  = (__bf16*)(ws + 37748736);   // 16 MB: attn out [M x 512]
  __bf16* WfT = (__bf16*)(ws + 54525952);
  __bf16* WgT = (__bf16*)(ws + 54591488);
  __bf16* WhT = (__bf16*)(ws + 54657024);
  __bf16* WvT = (__bf16*)(ws + 55181312);   // total ~53.1 MB

  const int TB = 256;  // 8 wave32s per block

  // Convert / transpose
  sa_cvt_x<<<(MTOT * CC / 4 + TB - 1) / TB, TB, 0, stream>>>(x, xb, MTOT * CC / 4);
  sa_cvt_wT<<<(CC * CD + TB - 1) / TB, TB, 0, stream>>>(Wf, WfT, CC, CD);
  sa_cvt_wT<<<(CC * CD + TB - 1) / TB, TB, 0, stream>>>(Wg, WgT, CC, CD);
  sa_cvt_wT<<<(CC * CC + TB - 1) / TB, TB, 0, stream>>>(Wh, WhT, CC, CC);
  sa_cvt_wT<<<(CC * CC + TB - 1) / TB, TB, 0, stream>>>(Wv, WvT, CC, CC);

  // Projections (WMMA)
  sa_gemm_proj<<<128, TB, 0, stream>>>(xb, WfT, fb, MTOT, CD, CC, 0);
  sa_gemm_proj<<<128, TB, 0, stream>>>(xb, WgT, gb, MTOT, CD, CC, 0);
  sa_gemm_proj<<<1024, TB, 0, stream>>>(xb, WhT, hvT, MTOT, CC, CC, NN);

  // Attention: softmax scores (writes beta to d_out), then o = beta @ hv
  sa_attn_beta<<<128, TB, 0, stream>>>(gb, fb, beta);
  sa_attn_o<<<1024, TB, 0, stream>>>(beta, hvT, ob);

  // Final 1x1 conv + residual
  sa_gemm_final<<<1024, TB, 0, stream>>>(ob, WvT, x, gamma, y);
}
